// LstmModel_41901700939843
// MI455X (gfx1250) — compile-verified, matrix operand order checked by reading, not run on hardware
//
#include <hip/hip_runtime.h>
#include <hip/hip_bf16.h>

typedef __attribute__((ext_vector_type(16))) _Float16 v16h;
typedef __attribute__((ext_vector_type(8)))  float    v8f;

#define BATCH 512
#define SEQ   512
#define NIN   51
#define HID   96
#define NOUT  51
#define OSTRIDE ((SEQ + 1) * NOUT)   // 513*51 per batch row

// K index inside a 32-wide chunk for element j of an f16 A/B fragment.
// ISA 7.12.2: lanes 0-15 (half=0): VGPR0..3 -> K 0..7, VGPR4..7 -> K 16..23
//             lanes 16-31 (half=1): K 8..15 and K 24..31.
__device__ __forceinline__ int kmap(int j, int half) {
    return (j < 8) ? (8 * half + j) : (16 + 8 * half + (j - 8));
}

__device__ __forceinline__ v16h zero16h() {
    v16h z;
#pragma unroll
    for (int j = 0; j < 16; ++j) z[j] = (_Float16)0.f;
    return z;
}

__device__ __forceinline__ v8f zero8f() {
    v8f z;
#pragma unroll
    for (int e = 0; e < 8; ++e) z[e] = 0.f;
    return z;
}

// Branch-free activations on hardware transcendentals (v_exp_f32 / v_rcp_f32).
__device__ __forceinline__ float fast_sigmoid(float x) {
    return __builtin_amdgcn_rcpf(1.f + __builtin_amdgcn_exp2f(-1.44269504f * x));
}
__device__ __forceinline__ float fast_tanh(float x) {
    // tanh(x) = 1 - 2/(1 + e^{2x});  e^{2x} = exp2(2*log2(e)*x)
    return 1.f - 2.f * __builtin_amdgcn_rcpf(1.f + __builtin_amdgcn_exp2f(2.88539009f * x));
}

// Load a B fragment (B = W^T, i.e. element (lane n, k) = W[n][k]) from f32
// global weights [nrows x ncols] row-major, with zero padding outside.
__device__ __forceinline__ v16h load_bfrag(const float* __restrict__ W,
                                           int nrows, int ncols,
                                           int nt, int kc, int lane) {
    v16h b;
    const int n    = nt * 16 + (lane & 15);
    const int half = lane >> 4;
#pragma unroll
    for (int j = 0; j < 16; ++j) {
        const int k = kc * 32 + kmap(j, half);
        float v = 0.f;
        if (n < nrows && k < ncols) v = W[(size_t)n * ncols + k];
        b[j] = (_Float16)v;
    }
    return b;
}

// Build an A fragment (M=16 rows) from an LDS f16 tile [16][stride].
__device__ __forceinline__ v16h load_afrag_lds(const _Float16* __restrict__ X,
                                               int stride, int kc, int lane) {
    v16h a;
    const int m    = lane & 15;
    const int half = lane >> 4;
#pragma unroll
    for (int j = 0; j < 16; ++j) {
        const int k = kc * 32 + kmap(j, half);
        a[j] = X[m * stride + k];
    }
    return a;
}

// ---------------------------------------------------------------------------
// Layer 0: x[B,S,51] (f32) -> h1[B,S,96] (f16).  One block per 16 batch rows,
// 6 waves; wave w owns hidden cols [16w,16w+16) of every gate.  Weights live
// in VGPRs for all 512 steps.
// ---------------------------------------------------------------------------
__global__ __launch_bounds__(192) void lstm_layer0(
    const float* __restrict__ x,
    const float* __restrict__ Wih, const float* __restrict__ Whh,
    const float* __restrict__ bih, const float* __restrict__ bhh,
    _Float16* __restrict__ h1) {
    __shared__ __align__(16) _Float16 xt[16][64];     // x tile, K padded 51 -> 64
    __shared__ __align__(16) _Float16 hlds[16][HID];  // current h (f16)

    const int tid  = threadIdx.x;
    const int lane = tid & 31;
    const int w    = tid >> 5;          // wave 0..5 -> hidden tile
    const int col  = lane & 15;
    const int half = lane >> 4;
    const int m0   = blockIdx.x * 16;   // batch row base

    // Per-wave weight fragments: 4 gates x (2 K-chunks Wih + 3 K-chunks Whh).
    v16h Bih[4][2], Bhh[4][3];
    float bias[4];
#pragma unroll
    for (int g = 0; g < 4; ++g) {
        const int nt = g * 6 + w;       // N-tile inside 384 gate columns
#pragma unroll
        for (int kc = 0; kc < 2; ++kc) Bih[g][kc] = load_bfrag(Wih, 384, NIN, nt, kc, lane);
#pragma unroll
        for (int kc = 0; kc < 3; ++kc) Bhh[g][kc] = load_bfrag(Whh, 384, HID, nt, kc, lane);
        const int n = nt * 16 + col;
        bias[g] = bih[n] + bhh[n];
    }

    for (int e = tid; e < 16 * HID; e += 192) (&hlds[0][0])[e] = (_Float16)0.f;
    __syncthreads();

    v16h ha[3];
#pragma unroll
    for (int kc = 0; kc < 3; ++kc) ha[kc] = zero16h();
    v8f c = zero8f();

    // h1 tile copy-out: one uint4 (8 halfs) per thread, 192 threads = 16x96.
    const int cr = tid / 12;            // row 0..15
    const int cc = (tid % 12) * 8;      // half-column 0,8,...,88

    for (int t = 0; t < SEQ; ++t) {
        // Stage x tile (f32 -> f16, zero-pad K to 64) + prefetch next step.
        for (int e = tid; e < 16 * 64; e += 192) {
            const int r = e >> 6, d = e & 63;
            float v = 0.f;
            if (d < NIN) {
                const size_t base = (size_t)(m0 + r) * SEQ * NIN + (size_t)t * NIN + d;
                v = x[base];
                if (t + 1 < SEQ) __builtin_prefetch(&x[base + NIN], 0, 1);
            }
            xt[r][d] = (_Float16)v;
        }
        __syncthreads();

        v16h xa[2];
#pragma unroll
        for (int kc = 0; kc < 2; ++kc) xa[kc] = load_afrag_lds(&xt[0][0], 64, kc, lane);

        v8f acc[4];
#pragma unroll
        for (int g = 0; g < 4; ++g) {
            v8f a;
#pragma unroll
            for (int e = 0; e < 8; ++e) a[e] = bias[g];
#pragma unroll
            for (int kc = 0; kc < 2; ++kc)
                a = __builtin_amdgcn_wmma_f32_16x16x32_f16(false, xa[kc], false, Bih[g][kc],
                                                           (short)0, a, false, false);
#pragma unroll
            for (int kc = 0; kc < 3; ++kc)
                a = __builtin_amdgcn_wmma_f32_16x16x32_f16(false, ha[kc], false, Bhh[g][kc],
                                                           (short)0, a, false, false);
            acc[g] = a;
        }

        // Gate activations + state update (lane-local in C layout).
#pragma unroll
        for (int e = 0; e < 8; ++e) {
            const float iv = fast_sigmoid(acc[0][e]);
            const float fv = fast_sigmoid(acc[1][e]);
            const float gv = fast_tanh(acc[2][e]);
            const float ov = fast_sigmoid(acc[3][e]);
            const float cn = fv * c[e] + iv * gv;
            c[e] = cn;
            hlds[e + 8 * half][w * 16 + col] = (_Float16)(ov * fast_tanh(cn));
        }
        __syncthreads();
#pragma unroll
        for (int kc = 0; kc < 3; ++kc) ha[kc] = load_afrag_lds(&hlds[0][0], HID, kc, lane);

        // Coalesced 128-bit copy of the h tile to global h1.
        {
            const uint4 v = *(const uint4*)&hlds[cr][cc];
            *(uint4*)&h1[(size_t)(m0 + cr) * SEQ * HID + (size_t)t * HID + cc] = v;
        }
        __syncthreads();  // hlds/xt reads done before next iteration rewrites
    }
}

// ---------------------------------------------------------------------------
// Layer 1 + fused output heads: h1 -> h2, y[b,t,:] = h2 @ Wo^T + bo stored
// per timestep, y_n = h_final @ Wn^T + bn stored at row S.
// ---------------------------------------------------------------------------
__global__ __launch_bounds__(192) void lstm_layer1_out(
    const _Float16* __restrict__ h1,
    const float* __restrict__ Wih, const float* __restrict__ Whh,
    const float* __restrict__ bih, const float* __restrict__ bhh,
    const float* __restrict__ Wo, const float* __restrict__ bo,
    const float* __restrict__ Wn, const float* __restrict__ bn,
    float* __restrict__ out) {
    __shared__ __align__(16) _Float16 xt[16][HID];
    __shared__ __align__(16) _Float16 hlds[16][HID];

    const int tid  = threadIdx.x;
    const int lane = tid & 31;
    const int w    = tid >> 5;
    const int col  = lane & 15;
    const int half = lane >> 4;
    const int m0   = blockIdx.x * 16;

    v16h Bih[4][3], Bhh[4][3];
    float bias[4];
#pragma unroll
    for (int g = 0; g < 4; ++g) {
        const int nt = g * 6 + w;
#pragma unroll
        for (int kc = 0; kc < 3; ++kc) {
            Bih[g][kc] = load_bfrag(Wih, 384, HID, nt, kc, lane);
            Bhh[g][kc] = load_bfrag(Whh, 384, HID, nt, kc, lane);
        }
        const int n = nt * 16 + col;
        bias[g] = bih[n] + bhh[n];
    }

    // Output-head weights: waves 0..3 each own one 16-wide N tile (pad 51->64).
    v16h WoF[3], WnF[3];
    float bov = 0.f, bnv = 0.f;
    if (w < 4) {
#pragma unroll
        for (int kc = 0; kc < 3; ++kc) {
            WoF[kc] = load_bfrag(Wo, NOUT, HID, w, kc, lane);
            WnF[kc] = load_bfrag(Wn, NOUT, HID, w, kc, lane);
        }
        const int oc = w * 16 + col;
        if (oc < NOUT) { bov = bo[oc]; bnv = bn[oc]; }
    }

    for (int e = tid; e < 16 * HID; e += 192) (&hlds[0][0])[e] = (_Float16)0.f;
    __syncthreads();

    v16h ha[3];
#pragma unroll
    for (int kc = 0; kc < 3; ++kc) ha[kc] = zero16h();
    v8f c = zero8f();

    // h1 tile copy-in: one uint4 (8 halfs) per thread.
    const int cr = tid / 12;
    const int cc = (tid % 12) * 8;

    for (int t = 0; t < SEQ; ++t) {
        // Coalesced 128-bit stage of the h1 tile.
        {
            const size_t base = (size_t)(m0 + cr) * SEQ * HID + (size_t)t * HID + cc;
            *(uint4*)&xt[cr][cc] = *(const uint4*)&h1[base];
            if (t + 1 < SEQ) __builtin_prefetch(&h1[base + HID], 0, 1);
        }
        __syncthreads();

        v16h xa[3];
#pragma unroll
        for (int kc = 0; kc < 3; ++kc) xa[kc] = load_afrag_lds(&xt[0][0], HID, kc, lane);

        v8f acc[4];
#pragma unroll
        for (int g = 0; g < 4; ++g) {
            v8f a;
#pragma unroll
            for (int e = 0; e < 8; ++e) a[e] = bias[g];
#pragma unroll
            for (int kc = 0; kc < 3; ++kc) {
                a = __builtin_amdgcn_wmma_f32_16x16x32_f16(false, xa[kc], false, Bih[g][kc],
                                                           (short)0, a, false, false);
                a = __builtin_amdgcn_wmma_f32_16x16x32_f16(false, ha[kc], false, Bhh[g][kc],
                                                           (short)0, a, false, false);
            }
            acc[g] = a;
        }

#pragma unroll
        for (int e = 0; e < 8; ++e) {
            const float iv = fast_sigmoid(acc[0][e]);
            const float fv = fast_sigmoid(acc[1][e]);
            const float gv = fast_tanh(acc[2][e]);
            const float ov = fast_sigmoid(acc[3][e]);
            const float cn = fv * c[e] + iv * gv;
            c[e] = cn;
            hlds[e + 8 * half][w * 16 + col] = (_Float16)(ov * fast_tanh(cn));
        }
        __syncthreads();
#pragma unroll
        for (int kc = 0; kc < 3; ++kc) ha[kc] = load_afrag_lds(&hlds[0][0], HID, kc, lane);

        // Fused per-timestep output projection (waves 0..3, wave-uniform).
        if (w < 4) {
            v8f y = zero8f();
#pragma unroll
            for (int kc = 0; kc < 3; ++kc)
                y = __builtin_amdgcn_wmma_f32_16x16x32_f16(false, ha[kc], false, WoF[kc],
                                                           (short)0, y, false, false);
            const int oc = w * 16 + col;
            if (oc < NOUT) {
#pragma unroll
                for (int e = 0; e < 8; ++e) {
                    const int row = e + 8 * half;
                    out[(size_t)(m0 + row) * OSTRIDE + (size_t)t * NOUT + oc] = y[e] + bov;
                }
            }
        }
        __syncthreads();  // hlds/xt consumed; safe to rewrite next iteration
    }

    // Final head: y_n = h_S @ Wn^T + bn at sequence row S.
    if (w < 4) {
        v8f y = zero8f();
#pragma unroll
        for (int kc = 0; kc < 3; ++kc)
            y = __builtin_amdgcn_wmma_f32_16x16x32_f16(false, ha[kc], false, WnF[kc],
                                                       (short)0, y, false, false);
        const int oc = w * 16 + col;
        if (oc < NOUT) {
#pragma unroll
            for (int e = 0; e < 8; ++e) {
                const int row = e + 8 * half;
                out[(size_t)(m0 + row) * OSTRIDE + (size_t)SEQ * NOUT + oc] = y[e] + bnv;
            }
        }
    }
}

extern "C" void kernel_launch(void* const* d_in, const int* in_sizes, int n_in,
                              void* d_out, int out_size, void* d_ws, size_t ws_size,
                              hipStream_t stream) {
    (void)in_sizes; (void)n_in; (void)out_size; (void)ws_size;
    const float* x    = (const float*)d_in[0];
    const float* Wih0 = (const float*)d_in[1];
    const float* Whh0 = (const float*)d_in[2];
    const float* bih0 = (const float*)d_in[3];
    const float* bhh0 = (const float*)d_in[4];
    const float* Wih1 = (const float*)d_in[5];
    const float* Whh1 = (const float*)d_in[6];
    const float* bih1 = (const float*)d_in[7];
    const float* bhh1 = (const float*)d_in[8];
    const float* Wo   = (const float*)d_in[9];
    const float* bo   = (const float*)d_in[10];
    const float* Wn   = (const float*)d_in[11];
    const float* bn   = (const float*)d_in[12];

    _Float16* h1 = (_Float16*)d_ws;  // B*S*H f16 = ~50.3 MB scratch

    lstm_layer0<<<BATCH / 16, 192, 0, stream>>>(x, Wih0, Whh0, bih0, bhh0, h1);
    lstm_layer1_out<<<BATCH / 16, 192, 0, stream>>>(h1, Wih1, Whh1, bih1, bhh1,
                                                    Wo, bo, Wn, bn, (float*)d_out);
}
